// ContinualCLora_3221225472407
// MI455X (gfx1250) — matrix operand order repaired and checked by reading, not compiled, not run on hardware
//
#include <hip/hip_runtime.h>

#define NUM_TASKS 5
#define POOL      5
#define R_RANK    8
#define TOPK      3
#define B_DIM     4
#define S_DIM     4096
#define DIN       1024
#define DOUT      1024
#define ROWS      (B_DIM * S_DIM)   // 16384
#define KPAD      32                // padded concat rank (TOPK*R=24 -> 32)

typedef __attribute__((ext_vector_type(16))) __bf16       bf16x16;
typedef __attribute__((ext_vector_type(8)))  float        floatx8;
typedef __attribute__((ext_vector_type(4)))  float        f32x4;
typedef __attribute__((ext_vector_type(2)))  unsigned int u32x2;
typedef __attribute__((ext_vector_type(4)))  unsigned int u32x4;
typedef __attribute__((ext_vector_type(8)))  unsigned int u32x8;

// ---------------- workspace layout (bytes) ----------------
#define V_OFF    ((size_t)0)                                   // 1024 f32 colsum
#define ABF_OFF  ((size_t)4096)                                // 16384x1024 bf16
#define WBF_OFF  (ABF_OFF + (size_t)ROWS * DIN * 2)            // 1024x1024 bf16
#define DCAT_OFF (WBF_OFF + (size_t)DOUT * DIN * 2)            // 32x1024 bf16 (B-layout)
#define UCAT_OFF (DCAT_OFF + (size_t)KPAD * DIN * 2)           // 1024x32 bf16 (B-layout)
#define T_OFF    (UCAT_OFF + (size_t)DOUT * KPAD * 2)          // 16384x32 bf16 (A-layout)

// ---------------- helpers ----------------
__device__ inline unsigned short f2bf(float f) {
  unsigned int u = __float_as_uint(f);
  u += 0x7fffu + ((u >> 16) & 1u);            // round-to-nearest-even
  return (unsigned short)(u >> 16);
}
__device__ inline float bf2f(unsigned short h) {
  return __uint_as_float(((unsigned int)h) << 16);
}

// A fragment (16x32 bf16, row-major memory, row stride `stride` elements).
// ISA layout: lane m (0-15) holds row m, K 0-7 then 16-23; lane m+16 holds K 8-15 / 24-31.
__device__ inline bf16x16 load_fragA(const unsigned short* __restrict__ base,
                                     int row0, int stride, int k0, int lane) {
  int r = lane & 15, half = lane >> 4;
  const unsigned short* p = base + (size_t)(row0 + r) * stride + k0 + half * 8;
  u32x4 lo = *(const u32x4*)p;                // global_load_b128
  u32x4 hi = *(const u32x4*)(p + 16);         // global_load_b128
  u32x8 v = __builtin_shufflevector(lo, hi, 0, 1, 2, 3, 4, 5, 6, 7);
  return __builtin_bit_cast(bf16x16, v);
}

// B fragment (32x16 bf16; GEMM column n is k-contiguous in memory, column stride `stride`).
// ISA layout: lane n (0-15) holds column n, K 0-7 then 16-23; lane n+16 holds K 8-15 / 24-31.
__device__ inline bf16x16 load_fragB(const unsigned short* __restrict__ base,
                                     int col0, int stride, int k0, int lane) {
  int n = lane & 15, half = lane >> 4;
  const unsigned short* p = base + (size_t)(col0 + n) * stride + k0 + half * 8;
  u32x4 lo = *(const u32x4*)p;                // global_load_b128
  u32x4 hi = *(const u32x4*)(p + 16);         // global_load_b128
  u32x8 v = __builtin_shufflevector(lo, hi, 0, 1, 2, 3, 4, 5, 6, 7);
  return __builtin_bit_cast(bf16x16, v);
}

__device__ inline floatx8 wmma_bf16(bf16x16 a, bf16x16 b, floatx8 c) {
  return __builtin_amdgcn_wmma_f32_16x16x32_bf16(false, a, false, b, (short)0, c,
                                                 false, false);
}

// ---------------- kernels ----------------
__global__ __launch_bounds__(256) void k_convert_bf16(const float* __restrict__ src,
                                                      unsigned short* __restrict__ dst,
                                                      int n4) {
  int i = blockIdx.x * 256 + threadIdx.x;
  if (i >= n4) return;
  f32x4 f = *(const f32x4*)(src + (size_t)i * 4);
  u32x2 o;
  o.x = (unsigned int)f2bf(f.x) | ((unsigned int)f2bf(f.y) << 16);
  o.y = (unsigned int)f2bf(f.z) | ((unsigned int)f2bf(f.w) << 16);
  *(u32x2*)(dst + (size_t)i * 4) = o;
}

__global__ __launch_bounds__(256) void k_zero(float* __restrict__ p, int n) {
  int i = blockIdx.x * 256 + threadIdx.x;
  if (i < n) p[i] = 0.0f;
}

__global__ __launch_bounds__(256) void k_colsum(const unsigned short* __restrict__ Abf,
                                                float* __restrict__ v) {
  int d = blockIdx.x * 256 + threadIdx.x;
  int r0 = blockIdx.y * 256;
  float s = 0.0f;
  for (int r = 0; r < 256; ++r) s += bf2f(Abf[(size_t)(r0 + r) * DIN + d]);
  atomicAdd(&v[d], s);
}

__global__ __launch_bounds__(256) void k_route(const float* __restrict__ v,
                                               const float* __restrict__ lora_route,
                                               const float* __restrict__ lora_down,
                                               const float* __restrict__ lora_up,
                                               const int* __restrict__ task_id_p,
                                               unsigned short* __restrict__ DcatT,
                                               unsigned short* __restrict__ UcatT) {
  __shared__ float red[256];
  __shared__ float omg[POOL];
  __shared__ float gate_s[TOPK];
  __shared__ int   idx_s[TOPK];
  __shared__ int   k_sh;
  int tid = threadIdx.x;

  float part[POOL];
#pragma unroll
  for (int p = 0; p < POOL; ++p) part[p] = 0.0f;
  const float* route1 = lora_route + (size_t)1 * DIN * POOL;  // route = lora_route[1]
  for (int d = tid; d < DIN; d += 256) {
    float vd = v[d];
#pragma unroll
    for (int p = 0; p < POOL; ++p) part[p] += vd * route1[d * POOL + p];
  }
  for (int p = 0; p < POOL; ++p) {
    red[tid] = part[p];
    __syncthreads();
    for (int s = 128; s > 0; s >>= 1) {
      if (tid < s) red[tid] += red[tid + s];
      __syncthreads();
    }
    if (tid == 0) omg[p] = red[0] * (1.0f / (float)ROWS);  // mean over B and S
    __syncthreads();
  }

  if (tid == 0) {
    int t = *task_id_p;
    if (t > NUM_TASKS) t = NUM_TASKS;
    int k = (t < TOPK) ? t : TOPK;
    int tlen = (t > POOL - 1) ? (POOL - 1) : t;    // sliced = omegas[1 : t+1]
    if (k > tlen) k = tlen;
    float sv[NUM_TASKS];
    int   si[NUM_TASKS];
    for (int i = 0; i < tlen; ++i) { sv[i] = omg[1 + i]; si[i] = i; }
    for (int j = 0; j < k; ++j) {                  // partial selection sort (ties -> lowest idx)
      int m = j;
      for (int i = j + 1; i < tlen; ++i)
        if (sv[i] > sv[m]) m = i;
      float tv = sv[j]; sv[j] = sv[m]; sv[m] = tv;
      int   ti = si[j]; si[j] = si[m]; si[m] = ti;
    }
    float mx = -3.4e38f;
    for (int j = 0; j < k; ++j) mx = fmaxf(mx, sv[j]);
    float se = 0.0f;
    for (int j = 0; j < k; ++j) { sv[j] = __expf(sv[j] - mx); se += sv[j]; }
    for (int j = 0; j < TOPK; ++j) {
      gate_s[j] = (j < k) ? (sv[j] / se) : 0.0f;
      idx_s[j]  = (j < k) ? si[j] : 0;
    }
    k_sh = k;
  }
  __syncthreads();
  int k = k_sh;

  // DcatT[j][d]: concatenated lora_down[I] columns, GEMM-B layout (k=d contiguous per column j)
  for (int e = tid; e < KPAD * DIN; e += 256) {
    int j = e / DIN, d = e % DIN;
    float val = 0.0f;
    if (j < k * R_RANK) {
      int pi = idx_s[j / R_RANK];
      val = lora_down[((size_t)pi * DIN + d) * R_RANK + (j % R_RANK)];
    }
    DcatT[(size_t)j * DIN + d] = f2bf(val);
  }
  // UcatT[o][j]: gate-folded lora_up[I], GEMM-B layout (k=j contiguous per column o)
  for (int e = tid; e < DOUT * KPAD; e += 256) {
    int o = e / KPAD, j = e % KPAD;
    float val = 0.0f;
    if (j < k * R_RANK) {
      int pi = idx_s[j / R_RANK];
      val = gate_s[j / R_RANK] * lora_up[((size_t)pi * R_RANK + (j % R_RANK)) * DOUT + o];
    }
    UcatT[(size_t)o * KPAD + j] = f2bf(val);
  }
}

// T = A_bf @ Dcat  (16384 x 32), batch-masked, stored bf16 row-major (A-fragment layout).
__global__ __launch_bounds__(256) void k_lora_t(const unsigned short* __restrict__ Abf,
                                                const unsigned short* __restrict__ DcatT,
                                                unsigned short* __restrict__ T) {
  int tid = threadIdx.x, lane = tid & 31, wave = tid >> 5;
  int rBase = blockIdx.x * 512 + wave * 64;
  floatx8 acc[4][2];
  floatx8 z = {0.f, 0.f, 0.f, 0.f, 0.f, 0.f, 0.f, 0.f};
#pragma unroll
  for (int mt = 0; mt < 4; ++mt)
#pragma unroll
    for (int nt = 0; nt < 2; ++nt) acc[mt][nt] = z;

  for (int kb = 0; kb < DIN; kb += 32) {
    bf16x16 b0 = load_fragB(DcatT, 0, DIN, kb, lane);
    bf16x16 b1 = load_fragB(DcatT, 16, DIN, kb, lane);
#pragma unroll
    for (int mt = 0; mt < 4; ++mt) {
      bf16x16 a = load_fragA(Abf, rBase + mt * 16, DIN, kb, lane);
      acc[mt][0] = wmma_bf16(a, b0, acc[mt][0]);
      acc[mt][1] = wmma_bf16(a, b1, acc[mt][1]);
    }
  }
  int r = lane & 15, half = lane >> 4;
#pragma unroll
  for (int mt = 0; mt < 4; ++mt)
#pragma unroll
    for (int nt = 0; nt < 2; ++nt) {
      int col = nt * 16 + r;
#pragma unroll
      for (int i = 0; i < 8; ++i) {
        int row = rBase + mt * 16 + half * 8 + i;
        int b = row >> 12;                                   // S = 4096
        unsigned short val =
            (b >= (B_DIM / 2)) ? f2bf(acc[mt][nt][i]) : (unsigned short)0;
        T[(size_t)row * KPAD + col] = val;
      }
    }
}

// out = A_bf @ W_bf^T  (+ T @ Ucat fused as one extra K=32 step)
__global__ __launch_bounds__(256) void k_gemm(const unsigned short* __restrict__ Abf,
                                              const unsigned short* __restrict__ Wbf,
                                              const unsigned short* __restrict__ T,
                                              const unsigned short* __restrict__ UcatT,
                                              float* __restrict__ out) {
  int tid = threadIdx.x, lane = tid & 31, wave = tid >> 5;
  int mBase = blockIdx.x * 128 + (wave & 3) * 32;   // wave tile: 32 rows
  int nBase = blockIdx.y * 128 + (wave >> 2) * 64;  //            64 cols
  floatx8 acc[2][4];
  floatx8 z = {0.f, 0.f, 0.f, 0.f, 0.f, 0.f, 0.f, 0.f};
#pragma unroll
  for (int mt = 0; mt < 2; ++mt)
#pragma unroll
    for (int nt = 0; nt < 4; ++nt) acc[mt][nt] = z;

  for (int kb = 0; kb < DIN; kb += 32) {
    if (kb + 32 < DIN) {  // pull next K-tile toward the WGP (global_prefetch_b8)
      __builtin_prefetch(Abf + (size_t)(mBase + (lane & 15)) * DIN + kb + 32, 0, 0);
      __builtin_prefetch(Wbf + (size_t)(nBase + (lane & 15)) * DIN + kb + 32, 0, 0);
    }
    bf16x16 a0 = load_fragA(Abf, mBase, DIN, kb, lane);
    bf16x16 a1 = load_fragA(Abf, mBase + 16, DIN, kb, lane);
#pragma unroll
    for (int nt = 0; nt < 4; ++nt) {
      bf16x16 b = load_fragB(Wbf, nBase + nt * 16, DIN, kb, lane);
      acc[0][nt] = wmma_bf16(a0, b, acc[0][nt]);
      acc[1][nt] = wmma_bf16(a1, b, acc[1][nt]);
    }
  }
  {  // LoRA delta: rank-24 (padded 32) correction as a final K-step
    bf16x16 a0 = load_fragA(T, mBase, KPAD, 0, lane);
    bf16x16 a1 = load_fragA(T, mBase + 16, KPAD, 0, lane);
#pragma unroll
    for (int nt = 0; nt < 4; ++nt) {
      bf16x16 b = load_fragB(UcatT, nBase + nt * 16, KPAD, 0, lane);
      acc[0][nt] = wmma_bf16(a0, b, acc[0][nt]);
      acc[1][nt] = wmma_bf16(a1, b, acc[1][nt]);
    }
  }
  int r = lane & 15, half = lane >> 4;
#pragma unroll
  for (int mt = 0; mt < 2; ++mt)
#pragma unroll
    for (int nt = 0; nt < 4; ++nt) {
      int col = nBase + nt * 16 + r;
      float* op = out + (size_t)(mBase + mt * 16 + half * 8) * DOUT + col;
#pragma unroll
      for (int i = 0; i < 8; ++i) op[(size_t)i * DOUT] = acc[mt][nt][i];
    }
}

// ---------------- launcher ----------------
extern "C" void kernel_launch(void* const* d_in, const int* in_sizes, int n_in,
                              void* d_out, int out_size, void* d_ws, size_t ws_size,
                              hipStream_t stream) {
  const float* input      = (const float*)d_in[0];
  const float* W          = (const float*)d_in[1];
  const float* lora_down  = (const float*)d_in[2];
  const float* lora_up    = (const float*)d_in[3];
  const float* lora_route = (const float*)d_in[4];
  const int*   task_id    = (const int*)d_in[5];
  float* out = (float*)d_out;

  char* ws = (char*)d_ws;
  float*          v     = (float*)(ws + V_OFF);
  unsigned short* Abf   = (unsigned short*)(ws + ABF_OFF);
  unsigned short* Wbf   = (unsigned short*)(ws + WBF_OFF);
  unsigned short* DcatT = (unsigned short*)(ws + DCAT_OFF);
  unsigned short* UcatT = (unsigned short*)(ws + UCAT_OFF);
  unsigned short* T     = (unsigned short*)(ws + T_OFF);

  k_convert_bf16<<<(ROWS * DIN / 4) / 256, 256, 0, stream>>>(input, Abf, ROWS * DIN / 4);
  k_convert_bf16<<<(DOUT * DIN / 4) / 256, 256, 0, stream>>>(W, Wbf, DOUT * DIN / 4);
  k_zero<<<4, 256, 0, stream>>>(v, DIN);
  k_colsum<<<dim3(DIN / 256, ROWS / 256), 256, 0, stream>>>(Abf, v);
  k_route<<<1, 256, 0, stream>>>(v, lora_route, lora_down, lora_up, task_id, DcatT, UcatT);
  k_lora_t<<<ROWS / 512, 256, 0, stream>>>(Abf, DcatT, T);
  k_gemm<<<dim3(ROWS / 128, DOUT / 128), 256, 0, stream>>>(Abf, Wbf, T, UcatT, out);
}
